// FusedSparseMoE_18451179504174
// MI455X (gfx1250) — matrix operand order repaired and checked by reading, not compiled, not run on hardware
//
#include <hip/hip_runtime.h>
#include <hip/hip_bf16.h>

// ---------------------------------------------------------------------------
// Fused sparse MoE for MI455X (gfx1250, wave32, WMMA bf16)
//   T=2048 tokens, D=1024, F=2048, E=8 experts, top-2 routing.
// Sparse schedule (top-2 only): 51.5 GFLOP, ~192MB fp32 weight traffic
// (L2-resident across M-tiles). bf16 WMMA w/ fp32 accum; weights converted
// fp32->bf16 on the fly in LDS; A tiles staged with async global->LDS loads.
// ---------------------------------------------------------------------------

#define T_TOK 2048
#define D_MODEL 1024
#define F_FF 2048
#define N_EXP 8
#define N_PAIR (T_TOK * 2)

typedef __attribute__((ext_vector_type(16))) __bf16 v16bf;
typedef __attribute__((ext_vector_type(8)))  float  v8f;

union FragU { uint4 u[2]; v16bf v; };

__device__ __forceinline__ unsigned pack_bf16(float lo, float hi) {
  union { __bf16 h[2]; unsigned u; } r;
  r.h[0] = (__bf16)lo;
  r.h[1] = (__bf16)hi;
  return r.u;
}

__device__ __forceinline__ float fast_silu(float g) {
  // g * sigmoid(g) with v_rcp_f32 instead of IEEE division.
  return g * __builtin_amdgcn_rcpf(1.f + __expf(-g));
}

// Async global -> LDS copy, 16 bytes per lane (gfx1250, tracked by ASYNCcnt).
__device__ __forceinline__ void async_load_b128(void* lds_dst,
                                                const void* gsrc) {
  unsigned lds = (unsigned)(size_t)lds_dst;
  unsigned long long g = (unsigned long long)gsrc;
  asm volatile("global_load_async_to_lds_b128 %0, %1, off"
               :
               : "v"(lds), "v"(g)
               : "memory");
}
__device__ __forceinline__ void wait_asynccnt0() {
  asm volatile("s_wait_asynccnt 0" ::: "memory");
}

// ---------------------------------------------------------------------------
// Kernel 1: router. One wave per token.
// ---------------------------------------------------------------------------
__global__ __launch_bounds__(256) void moe_router_kernel(
    const float* __restrict__ x, const float* __restrict__ rw,
    int* __restrict__ topk_idx, float* __restrict__ topk_w) {
  int tok  = (blockIdx.x * blockDim.x + threadIdx.x) >> 5;
  int lane = threadIdx.x & 31;
  if (tok >= T_TOK) return;
  const float* xr = x + (size_t)tok * D_MODEL;
  float acc[N_EXP];
#pragma unroll
  for (int e = 0; e < N_EXP; ++e) acc[e] = 0.f;
  for (int d = lane; d < D_MODEL; d += 32) {
    float xv = xr[d];
    const float* rwd = rw + (size_t)d * N_EXP;
#pragma unroll
    for (int e = 0; e < N_EXP; ++e) acc[e] += xv * rwd[e];
  }
#pragma unroll
  for (int e = 0; e < N_EXP; ++e) {
#pragma unroll
    for (int off = 16; off > 0; off >>= 1)
      acc[e] += __shfl_xor(acc[e], off, 32);
  }
  if (lane == 0) {
    float mx = acc[0];
#pragma unroll
    for (int e = 1; e < N_EXP; ++e) mx = fmaxf(mx, acc[e]);
    float pe[N_EXP];
#pragma unroll
    for (int e = 0; e < N_EXP; ++e) pe[e] = __expf(acc[e] - mx);
    int i0 = 0;
#pragma unroll
    for (int e = 1; e < N_EXP; ++e) if (pe[e] > pe[i0]) i0 = e;
    int i1 = (i0 == 0) ? 1 : 0;
#pragma unroll
    for (int e = 0; e < N_EXP; ++e)
      if (e != i0 && pe[e] > pe[i1]) i1 = e;
    float w0 = pe[i0], w1 = pe[i1];
    float inv = __builtin_amdgcn_rcpf(w0 + w1);
    topk_idx[tok * 2 + 0] = i0;
    topk_idx[tok * 2 + 1] = i1;
    topk_w[tok * 2 + 0] = w0 * inv;
    topk_w[tok * 2 + 1] = w1 * inv;
  }
}

__global__ void moe_zero_counts_kernel(int* __restrict__ counts) {
  if (threadIdx.x < N_EXP) counts[threadIdx.x] = 0;
}

// Binning: list order nondeterministic but numerically irrelevant (each pair
// row computed independently, stored exactly once).
__global__ void moe_bin_kernel(const int* __restrict__ topk_idx,
                               int* __restrict__ counts,
                               int* __restrict__ rowlist) {
  int p = blockIdx.x * blockDim.x + threadIdx.x;
  if (p >= N_PAIR) return;
  int e = topk_idx[p];
  int pos = atomicAdd(&counts[e], 1);
  rowlist[e * T_TOK + pos] = p;
}

__global__ __launch_bounds__(256) void moe_cvt_x_kernel(
    const float* __restrict__ x, __bf16* __restrict__ xbf) {
  int i = (blockIdx.x * blockDim.x + threadIdx.x) * 4;
  if (i >= T_TOK * D_MODEL) return;
  float4 v = *(const float4*)(x + i);
  uint2 o;
  o.x = pack_bf16(v.x, v.y);
  o.y = pack_bf16(v.z, v.w);
  *(uint2*)(xbf + i) = o;
}

// ---------------------------------------------------------------------------
// WMMA fragment helpers (CDNA5 ISA 7.12.2, wave32).
// A (16x32): lane<16 -> M=lane,    K=[0..7],[16..23]
//            lane>=16 -> M=lane-16, K=[8..15],[24..31]
// B (32x16, LDS N-major): lane<16 -> N=lane, K=[0..7],[16..23]; lane>=16 -> K=[8..15],[24..31]
// ---------------------------------------------------------------------------
__device__ __forceinline__ v16bf load_frag_a(const __bf16* rowptr, int lane) {
  int kh = (lane < 16) ? 0 : 8;
  FragU f;
  f.u[0] = *(const uint4*)(rowptr + kh);
  f.u[1] = *(const uint4*)(rowptr + kh + 16);
  return f.v;
}
__device__ __forceinline__ v16bf load_frag_b(const __bf16* colptr, int lane) {
  int kb = (lane < 16) ? 0 : 8;
  FragU f;
  f.u[0] = *(const uint4*)(colptr + kb);
  f.u[1] = *(const uint4*)(colptr + kb + 16);
  return f.v;
}

// ---------------------------------------------------------------------------
// Kernel 4: GEMM1 — H = silu(X W1) * (X W2). Tile M=64 x N=64, 8 waves
// (4 m-blocks x 2 n-halves); 4 WMMA per wave per K-step (W1+W2 share A frag).
// ---------------------------------------------------------------------------
__global__ __launch_bounds__(256) void moe_gemm1_kernel(
    const __bf16* __restrict__ xbf,
    const float* __restrict__ w1, const float* __restrict__ w2,
    const int* __restrict__ counts, const int* __restrict__ rowlist,
    __bf16* __restrict__ hbuf) {
  int e = blockIdx.z;
  int n_e = counts[e];
  int m0 = blockIdx.y * 64;
  if (m0 >= n_e) return;
  int f0 = blockIdx.x * 64;
  const float* w1e = w1 + (size_t)e * D_MODEL * F_FF + f0;
  const float* w2e = w2 + (size_t)e * D_MODEL * F_FF + f0;

  __shared__ __align__(16) __bf16 As[64][32];   // [m][k]
  __shared__ __align__(16) __bf16 B1s[64][32];  // [n][k] (N-major)
  __shared__ __align__(16) __bf16 B2s[64][32];
  __shared__ int rows_s[64];

  int tid   = threadIdx.x;
  int wave  = tid >> 5;
  int lane  = tid & 31;
  int mblk  = wave & 3;   // 0..3 -> M offset *16
  int nhalf = wave >> 2;  // 0..1 -> N offset *32

  if (tid < 64) {
    int r = m0 + tid;
    rows_s[tid] = (r < n_e) ? rowlist[e * T_TOK + r] : -1;
  }
  __syncthreads();

  // A-tile: thread -> (row am, 8 k's at ak), filled via async global->LDS.
  int am = tid >> 2;
  int ak = (tid & 3) * 8;
  int p_a = rows_s[am];
  const __bf16* asrc =
      xbf + (size_t)((p_a >= 0) ? (p_a >> 1) : 0) * D_MODEL + ak;
  __bf16* adst = &As[am][ak];
  // B-load assignment: thread -> (k-pair bk, 4 n's at bn)
  int bn = (tid & 15) * 4;
  int bk = (tid >> 4) * 2;

  v8f acc1[2] = {v8f{}, v8f{}};
  v8f acc2[2] = {v8f{}, v8f{}};

  for (int k0 = 0; k0 < D_MODEL; k0 += 32) {
    if (k0 + 32 < D_MODEL) {
      __builtin_prefetch(w1e + (size_t)(k0 + 32 + bk) * F_FF + bn, 0, 3);
      __builtin_prefetch(w2e + (size_t)(k0 + 32 + bk) * F_FF + bn, 0, 3);
    }
    // Issue global loads before the barrier (overlap prior WMMAs).
    float4 r10 = *(const float4*)(w1e + (size_t)(k0 + bk) * F_FF + bn);
    float4 r11 = *(const float4*)(w1e + (size_t)(k0 + bk + 1) * F_FF + bn);
    float4 r20 = *(const float4*)(w2e + (size_t)(k0 + bk) * F_FF + bn);
    float4 r21 = *(const float4*)(w2e + (size_t)(k0 + bk + 1) * F_FF + bn);
    __syncthreads();  // prior iteration's fragment reads complete
    async_load_b128(adst, asrc + k0);  // A tile: 256 lanes x 16B = 4KB
    {
      const float* a0 = &r10.x;
      const float* a1 = &r11.x;
      const float* c0 = &r20.x;
      const float* c1 = &r21.x;
#pragma unroll
      for (int j = 0; j < 4; ++j) {
        *(unsigned*)(&B1s[bn + j][bk]) = pack_bf16(a0[j], a1[j]);
        *(unsigned*)(&B2s[bn + j][bk]) = pack_bf16(c0[j], c1[j]);
      }
    }
    wait_asynccnt0();
    __syncthreads();

    v16bf af = load_frag_a(&As[mblk * 16 + (lane & 15)][0], lane);
    int col0 = (nhalf * 2 + 0) * 16 + (lane & 15);
    int col1 = (nhalf * 2 + 1) * 16 + (lane & 15);
    v16bf b1a = load_frag_b(&B1s[col0][0], lane);
    v16bf b2a = load_frag_b(&B2s[col0][0], lane);
    v16bf b1b = load_frag_b(&B1s[col1][0], lane);
    v16bf b2b = load_frag_b(&B2s[col1][0], lane);
    acc1[0] = __builtin_amdgcn_wmma_f32_16x16x32_bf16(
        false, af, false, b1a, (short)0, acc1[0], false, false);
    acc2[0] = __builtin_amdgcn_wmma_f32_16x16x32_bf16(
        false, af, false, b2a, (short)0, acc2[0], false, false);
    acc1[1] = __builtin_amdgcn_wmma_f32_16x16x32_bf16(
        false, af, false, b1b, (short)0, acc1[1], false, false);
    acc2[1] = __builtin_amdgcn_wmma_f32_16x16x32_bf16(
        false, af, false, b2b, (short)0, acc2[1], false, false);
  }

  // Epilogue: h = silu(a1) * a2 -> bf16 H rows (pair-indexed).
  int mbase = mblk * 16 + ((lane < 16) ? 0 : 8);
  if (m0 + 64 <= n_e) {
    // Full tile: no predication.
#pragma unroll
    for (int ns = 0; ns < 2; ++ns) {
      int nglob = f0 + (nhalf * 2 + ns) * 16 + (lane & 15);
#pragma unroll
      for (int v = 0; v < 8; ++v) {
        int p = rows_s[mbase + v];
        float h = fast_silu(acc1[ns][v]) * acc2[ns][v];
        hbuf[(size_t)p * F_FF + nglob] = (__bf16)h;
      }
    }
  } else {
#pragma unroll
    for (int ns = 0; ns < 2; ++ns) {
      int nglob = f0 + (nhalf * 2 + ns) * 16 + (lane & 15);
#pragma unroll
      for (int v = 0; v < 8; ++v) {
        int p = rows_s[mbase + v];
        if (p >= 0) {
          float h = fast_silu(acc1[ns][v]) * acc2[ns][v];
          hbuf[(size_t)p * F_FF + nglob] = (__bf16)h;
        }
      }
    }
  }
}

// ---------------------------------------------------------------------------
// Kernel 5: GEMM2 — Y = H W3 (K=F=2048, N=D). Tile M=64 x N=64.
// ---------------------------------------------------------------------------
__global__ __launch_bounds__(256) void moe_gemm2_kernel(
    const __bf16* __restrict__ hbuf, const float* __restrict__ w3,
    const int* __restrict__ counts, const int* __restrict__ rowlist,
    float* __restrict__ ybuf) {
  int e = blockIdx.z;
  int n_e = counts[e];
  int m0 = blockIdx.y * 64;
  if (m0 >= n_e) return;
  int d0 = blockIdx.x * 64;
  const float* w3e = w3 + (size_t)e * F_FF * D_MODEL + d0;

  __shared__ __align__(16) __bf16 As[64][32];  // [m][k]
  __shared__ __align__(16) __bf16 Bs[64][32];  // [n][k]
  __shared__ int rows_s[64];

  int tid   = threadIdx.x;
  int wave  = tid >> 5;
  int lane  = tid & 31;
  int mblk  = wave & 3;
  int nhalf = wave >> 2;

  if (tid < 64) {
    int r = m0 + tid;
    rows_s[tid] = (r < n_e) ? rowlist[e * T_TOK + r] : -1;
  }
  __syncthreads();

  int am = tid >> 2;
  int ak = (tid & 3) * 8;
  int p_a = rows_s[am];
  const __bf16* asrc = hbuf + (size_t)((p_a >= 0) ? p_a : 0) * F_FF + ak;
  __bf16* adst = &As[am][ak];
  int bn = (tid & 15) * 4;
  int bk = (tid >> 4) * 2;

  v8f acc[2] = {v8f{}, v8f{}};

  for (int k0 = 0; k0 < F_FF; k0 += 32) {
    if (k0 + 32 < F_FF) {
      __builtin_prefetch(w3e + (size_t)(k0 + 32 + bk) * D_MODEL + bn, 0, 3);
    }
    float4 r0 = *(const float4*)(w3e + (size_t)(k0 + bk) * D_MODEL + bn);
    float4 r1 = *(const float4*)(w3e + (size_t)(k0 + bk + 1) * D_MODEL + bn);
    __syncthreads();
    async_load_b128(adst, asrc + k0);
    {
      const float* a0 = &r0.x;
      const float* a1 = &r1.x;
#pragma unroll
      for (int j = 0; j < 4; ++j)
        *(unsigned*)(&Bs[bn + j][bk]) = pack_bf16(a0[j], a1[j]);
    }
    wait_asynccnt0();
    __syncthreads();

    v16bf af = load_frag_a(&As[mblk * 16 + (lane & 15)][0], lane);
    int col0 = (nhalf * 2 + 0) * 16 + (lane & 15);
    int col1 = (nhalf * 2 + 1) * 16 + (lane & 15);
    v16bf ba = load_frag_b(&Bs[col0][0], lane);
    v16bf bb = load_frag_b(&Bs[col1][0], lane);
    acc[0] = __builtin_amdgcn_wmma_f32_16x16x32_bf16(
        false, af, false, ba, (short)0, acc[0], false, false);
    acc[1] = __builtin_amdgcn_wmma_f32_16x16x32_bf16(
        false, af, false, bb, (short)0, acc[1], false, false);
  }

  int mbase = mblk * 16 + ((lane < 16) ? 0 : 8);
  if (m0 + 64 <= n_e) {
#pragma unroll
    for (int ns = 0; ns < 2; ++ns) {
      int nglob = d0 + (nhalf * 2 + ns) * 16 + (lane & 15);
#pragma unroll
      for (int v = 0; v < 8; ++v)
        ybuf[(size_t)rows_s[mbase + v] * D_MODEL + nglob] = acc[ns][v];
    }
  } else {
#pragma unroll
    for (int ns = 0; ns < 2; ++ns) {
      int nglob = d0 + (nhalf * 2 + ns) * 16 + (lane & 15);
#pragma unroll
      for (int v = 0; v < 8; ++v) {
        int p = rows_s[mbase + v];
        if (p >= 0) ybuf[(size_t)p * D_MODEL + nglob] = acc[ns][v];
      }
    }
  }
}

// ---------------------------------------------------------------------------
// Kernel 6: combine — out[t] = w0*Y[2t] + w1*Y[2t+1] (float4 vectorized).
// ---------------------------------------------------------------------------
__global__ __launch_bounds__(256) void moe_combine_kernel(
    const float* __restrict__ ybuf, const float* __restrict__ topk_w,
    float* __restrict__ out) {
  int i = (blockIdx.x * blockDim.x + threadIdx.x) * 4;
  if (i >= T_TOK * D_MODEL) return;
  int t = i >> 10;  // / D_MODEL
  int d = i & (D_MODEL - 1);
  float w0 = topk_w[2 * t + 0];
  float w1 = topk_w[2 * t + 1];
  float4 y0 = *(const float4*)(ybuf + (size_t)(2 * t) * D_MODEL + d);
  float4 y1 = *(const float4*)(ybuf + (size_t)(2 * t + 1) * D_MODEL + d);
  float4 o;
  o.x = w0 * y0.x + w1 * y1.x;
  o.y = w0 * y0.y + w1 * y1.y;
  o.z = w0 * y0.z + w1 * y1.z;
  o.w = w0 * y0.w + w1 * y1.w;
  *(float4*)(out + i) = o;
}

// ---------------------------------------------------------------------------
// Launch. Workspace layout (~36.2 MB):
//   counts(256B) | rowlist 64KB | topk_idx 16KB | topk_w 16KB |
//   xbf 4MB | hbuf 16MB | ybuf 16MB
// ---------------------------------------------------------------------------
extern "C" void kernel_launch(void* const* d_in, const int* in_sizes, int n_in,
                              void* d_out, int out_size, void* d_ws,
                              size_t ws_size, hipStream_t stream) {
  (void)in_sizes; (void)n_in; (void)out_size; (void)ws_size;
  const float* x  = (const float*)d_in[0];
  const float* rw = (const float*)d_in[1];
  const float* w1 = (const float*)d_in[2];
  const float* w2 = (const float*)d_in[3];
  const float* w3 = (const float*)d_in[4];
  float* out = (float*)d_out;

  char* ws = (char*)d_ws;
  size_t off = 0;
  int* counts = (int*)(ws + off);           off += 256;
  int* rowlist = (int*)(ws + off);          off += (size_t)N_EXP * T_TOK * 4;
  int* topk_idx = (int*)(ws + off);         off += (size_t)N_PAIR * 4;
  float* topk_w = (float*)(ws + off);       off += (size_t)N_PAIR * 4;
  __bf16* xbf = (__bf16*)(ws + off);        off += (size_t)T_TOK * D_MODEL * 2;
  __bf16* hbuf = (__bf16*)(ws + off);       off += (size_t)N_PAIR * F_FF * 2;
  float* ybuf = (float*)(ws + off);         off += (size_t)N_PAIR * D_MODEL * 4;

  moe_router_kernel<<<T_TOK / 8, 256, 0, stream>>>(x, rw, topk_idx, topk_w);
  moe_zero_counts_kernel<<<1, 32, 0, stream>>>(counts);
  moe_bin_kernel<<<(N_PAIR + 255) / 256, 256, 0, stream>>>(topk_idx, counts,
                                                           rowlist);
  moe_cvt_x_kernel<<<(T_TOK * D_MODEL / 4 + 255) / 256, 256, 0, stream>>>(x,
                                                                          xbf);
  moe_gemm1_kernel<<<dim3(F_FF / 64, T_TOK / 64, N_EXP), 256, 0, stream>>>(
      xbf, w1, w2, counts, rowlist, hbuf);
  moe_gemm2_kernel<<<dim3(D_MODEL / 64, T_TOK / 64, N_EXP), 256, 0, stream>>>(
      hbuf, w3, counts, rowlist, ybuf);
  moe_combine_kernel<<<(T_TOK * D_MODEL / 4 + 255) / 256, 256, 0, stream>>>(
      ybuf, topk_w, out);
}
